// MoELayer_29626684408493
// MI455X (gfx1250) — matrix operand order repaired and checked by reading, not compiled, not run on hardware
//
#include <hip/hip_runtime.h>
#include <math.h>
#include <stdint.h>

// ---------------------------------------------------------------------------
// MoE transformer block for MI455X (gfx1250, wave32, WMMA).
// GEMMs: bf16 A/B, f32 accumulate via v_wmma_f32_16x16x32_bf16, double-buffered
// LDS with async global->LDS staging (ASYNCcnt) when the toolchain exposes it.
// ---------------------------------------------------------------------------

typedef __bf16 bf16;
typedef __attribute__((ext_vector_type(8)))  __bf16 v8bf;
typedef __attribute__((ext_vector_type(16))) __bf16 v16bf;
typedef __attribute__((ext_vector_type(8)))  float  v8f;
typedef __attribute__((ext_vector_type(4)))  int    v4i;

#define THREADS 256

#define AS1 __attribute__((address_space(1)))
#define AS3 __attribute__((address_space(3)))

#if defined(__has_builtin)
#if __has_builtin(__builtin_amdgcn_global_load_async_to_lds_b128) && \
    __has_builtin(__builtin_amdgcn_s_wait_asynccnt)
#define USE_ASYNC 1
#endif
#endif
#ifndef USE_ASYNC
#define USE_ASYNC 0
#endif

// ------------------------- reduction helpers (wave32) ----------------------
__device__ __forceinline__ float warp_sum(float v) {
    for (int o = 16; o; o >>= 1) v += __shfl_xor(v, o, 32);
    return v;
}
__device__ __forceinline__ float warp_max(float v) {
    for (int o = 16; o; o >>= 1) v = fmaxf(v, __shfl_xor(v, o, 32));
    return v;
}
__device__ __forceinline__ float block_sum(float v, float* s) {
    v = warp_sum(v);
    __syncthreads();
    if ((threadIdx.x & 31) == 0) s[threadIdx.x >> 5] = v;
    __syncthreads();
    float r = s[0];
    for (int i = 1; i < 8; ++i) r += s[i];
    return r;
}
__device__ __forceinline__ float block_max(float v, float* s) {
    v = warp_max(v);
    __syncthreads();
    if ((threadIdx.x & 31) == 0) s[threadIdx.x >> 5] = v;
    __syncthreads();
    float r = s[0];
    for (int i = 1; i < 8; ++i) r = fmaxf(r, s[i]);
    return r;
}
__device__ __forceinline__ float gelu_exact(float v) {
    return 0.5f * v * (1.0f + erff(v * 0.70710678118654752f));
}

// ------------------------------ tiny kernels -------------------------------
__global__ void k_cvt_bf16(const float* __restrict__ s, bf16* __restrict__ d, size_t n) {
    size_t i  = (size_t)blockIdx.x * blockDim.x + threadIdx.x;
    size_t st = (size_t)gridDim.x * blockDim.x;
    for (; i < n; i += st) d[i] = (bf16)s[i];
}

// LN (norm1) then lucidrains RMSNorm (moe_gamma) fused; one block per token.
__global__ __launch_bounds__(THREADS) void k_ln_rms(
    const float* __restrict__ x, const float* __restrict__ w,
    const float* __restrict__ b, const float* __restrict__ g,
    bf16* __restrict__ xn)
{
    __shared__ float red[8];
    const size_t row = blockIdx.x;
    const float* xr  = x + row * 1024;
    const int t = threadIdx.x;
    float v[4], s = 0.f, sq = 0.f;
    for (int i = 0; i < 4; ++i) { v[i] = xr[t + 256 * i]; s += v[i]; sq += v[i] * v[i]; }
    float mean = block_sum(s,  red) * (1.f / 1024.f);
    float msq  = block_sum(sq, red) * (1.f / 1024.f);
    float rstd = rsqrtf(msq - mean * mean + 1e-5f);
    float h[4], hs = 0.f;
    for (int i = 0; i < 4; ++i) {
        int c = t + 256 * i;
        h[i] = (v[i] - mean) * rstd * w[c] + b[c];
        hs += h[i] * h[i];
    }
    float nrm = sqrtf(block_sum(hs, red));
    float sc  = 32.f / fmaxf(nrm, 1e-12f);   // sqrt(1024)=32
    for (int i = 0; i < 4; ++i) {
        int c = t + 256 * i;
        xn[row * 1024 + c] = (bf16)(h[i] * sc * g[c]);
    }
}

// Plain LayerNorm -> bf16, one block per token (norm2).
__global__ __launch_bounds__(THREADS) void k_ln(
    const float* __restrict__ x, const float* __restrict__ w,
    const float* __restrict__ b, bf16* __restrict__ out)
{
    __shared__ float red[8];
    const size_t row = blockIdx.x;
    const float* xr  = x + row * 1024;
    const int t = threadIdx.x;
    float v[4], s = 0.f, sq = 0.f;
    for (int i = 0; i < 4; ++i) { v[i] = xr[t + 256 * i]; s += v[i]; sq += v[i] * v[i]; }
    float mean = block_sum(s,  red) * (1.f / 1024.f);
    float msq  = block_sum(sq, red) * (1.f / 1024.f);
    float rstd = rsqrtf(msq - mean * mean + 1e-5f);
    for (int i = 0; i < 4; ++i) {
        int c = t + 256 * i;
        out[row * 1024 + c] = (bf16)((v[i] - mean) * rstd * w[c] + b[c]);
    }
}

// xm[b,s,d] = mean over 8 consecutive tokens of xn.
__global__ void k_groupmean(const bf16* __restrict__ xn, bf16* __restrict__ xm, size_t total) {
    size_t i  = (size_t)blockIdx.x * blockDim.x + threadIdx.x;
    size_t st = (size_t)gridDim.x * blockDim.x;
    for (; i < total; i += st) {
        int d = (int)(i & 1023);
        size_t r = i >> 10;
        int s = (int)(r & 511);
        int b = (int)(r >> 9);
        const bf16* p = xn + (((size_t)b * 4096 + (size_t)s * 8) << 10) + d;
        float acc = 0.f;
        for (int e = 0; e < 8; ++e) acc += (float)p[(size_t)e << 10];
        xm[i] = (bf16)(acc * 0.125f);
    }
}

// RMS-normalize slot embeddings: sepre[b*512+s][e*1024+d] -> se[(b*8+e)*512+s][d]
__global__ __launch_bounds__(THREADS) void k_rms_slots(
    const float* __restrict__ sepre, const float* __restrict__ gamma,
    bf16* __restrict__ se)
{
    __shared__ float red[8];
    int blk = blockIdx.x;          // 16384 = b*512*8
    int bs = blk >> 3, e = blk & 7;
    int b = bs >> 9, s = bs & 511;
    const float* src = sepre + (size_t)bs * 8192 + (size_t)e * 1024;
    const int t = threadIdx.x;
    float v[4], sq = 0.f;
    for (int i = 0; i < 4; ++i) { v[i] = src[t + 256 * i]; sq += v[i] * v[i]; }
    float nrm = sqrtf(block_sum(sq, red));
    float sc  = 32.f / fmaxf(nrm, 1e-12f);
    bf16* dst = se + ((size_t)((b * 8 + e) * 512 + s)) * 1024;
    for (int i = 0; i < 4; ++i) {
        int c = t + 256 * i;
        dst[c] = (bf16)(v[i] * sc * gamma[c]);
    }
}

// dispatch: softmax over tokens (column softmax of logits[b][n][es]); writes
// transposed dispT[e][b][s][n] so slot accumulation becomes a plain NN GEMM.
__global__ __launch_bounds__(THREADS) void k_softmax_col(
    const float* __restrict__ logits, bf16* __restrict__ dispT)
{
    int es = blockIdx.x * THREADS + threadIdx.x;
    int b  = blockIdx.y;
    const float* col = logits + (size_t)b * 4096 * 4096 + es;
    float mx = -3.4e38f, sum = 0.f;
    for (int n = 0; n < 4096; ++n) {
        float v = col[(size_t)n << 12];
        if (v > mx) { sum *= expf(mx - v); mx = v; }
        sum += expf(v - mx);
    }
    float inv = 1.f / sum;
    int e = es >> 9, s = es & 511;
    bf16* out = dispT + (((size_t)((e * 4 + b) * 512 + s)) << 12);
    for (int n = 0; n < 4096; ++n)
        out[n] = (bf16)(expf(col[(size_t)n << 12] - mx) * inv);
}

// combine: row softmax over E*s = 4096; one block per (b,n) row.
__global__ __launch_bounds__(THREADS) void k_softmax_row(
    const float* __restrict__ logits, bf16* __restrict__ comb)
{
    __shared__ float red[8];
    size_t row = blockIdx.x;
    const float* src = logits + (row << 12);
    const int t = threadIdx.x;
    float v[16], mx = -3.4e38f;
    for (int i = 0; i < 16; ++i) { v[i] = src[t + 256 * i]; mx = fmaxf(mx, v[i]); }
    mx = block_max(mx, red);
    float s = 0.f;
    for (int i = 0; i < 16; ++i) { v[i] = expf(v[i] - mx); s += v[i]; }
    s = block_sum(s, red);
    float inv = 1.f / s;
    bf16* dst = comb + (row << 12);
    for (int i = 0; i < 16; ++i) dst[t + 256 * i] = (bf16)(v[i] * inv);
}

// GEGLU: act = a * gelu(g); act rows padded to 2736 cols so expert-2 A tiles
// stay 16B-aligned for async b128 staging.
__global__ void k_geglu(const bf16* __restrict__ hdn, bf16* __restrict__ act, size_t total) {
    size_t i  = (size_t)blockIdx.x * blockDim.x + threadIdx.x;
    size_t st = (size_t)gridDim.x * blockDim.x;
    for (; i < total; i += st) {
        size_t row = i / 2730;
        int j = (int)(i - row * 2730);
        const bf16* r = hdn + row * 5460;
        float a = (float)r[j];
        float g = (float)r[2730 + j];
        act[row * 2736 + j] = (bf16)(a * gelu_exact(g));
    }
}

// ------------------------------- GEMM core ---------------------------------
// C[M,N] = A[M,K] x B  (BLAY=0: B row-major [K][N]; BLAY=1: B row-major [N][K],
// i.e. C = A x B^T). 256 threads, 8 waves in 2x4, block tile 128x128x32.
// Wave tile 64x32 = 8 x v_wmma_f32_16x16x32_bf16 per K-step.
// Double-buffered LDS; next tile staged with GLOBAL_LOAD_ASYNC_TO_LDS_B128
// (ASYNCcnt + s_wait_asynccnt) while current tile's WMMAs run.
// EPI: 0=none, 1=+bias, 2=+bias then exact GELU.  RESID: += resid (f32).
// CPERM: scatter C row m -> (m>>9)*cp1 + z*cp2 + (m&511)  (expert->token map).
#define BM 128
#define BN 128
#define BK 32
#define LDSS 40   // padded K-stride in LDS (bf16 elements); 80B, 16B aligned

template<int BLAY, int EPI, bool RESID, bool CPERM, typename CT>
__global__ __launch_bounds__(THREADS) void k_gemm(
    const bf16* __restrict__ A, size_t strideA, int lda,
    const bf16* __restrict__ B, size_t strideB, int ldb, int bmod,
    CT* __restrict__ C, size_t strideC, int ldc,
    const float* __restrict__ bias, size_t strideBias,
    const float* __restrict__ resid, size_t strideR, int ldr,
    int M, int N, int K, int cp1, int cp2)
{
    __shared__ __align__(16) bf16 sA[2][BM * LDSS];   // [m][k]
    __shared__ __align__(16) bf16 sB[2][BN * LDSS];   // [n][k] (transposed)

    const int z  = blockIdx.z;
    const bf16* Az = A + (size_t)z * strideA;
    const bf16* Bz = B + (size_t)(z % bmod) * strideB;
    const int n0 = blockIdx.x * BN;
    const int m0 = blockIdx.y * BM;

    const int tid  = threadIdx.x;
    const int lane = tid & 31;
    const int wave = tid >> 5;
    const int wm = wave & 1;      // wave row   (2 along M -> 64 rows each)
    const int wn = wave >> 1;     // wave col   (4 along N -> 32 cols each)
    const int lm = lane & 15;
    const int lh = lane >> 4;

    v8f acc[4][2];
    for (int mi = 0; mi < 4; ++mi)
        for (int ni = 0; ni < 2; ++ni)
            for (int r = 0; r < 8; ++r) acc[mi][ni][r] = 0.f;

    const int ktiles = (K + BK - 1) / BK;

    // ---- tile staging (global -> LDS) ----
    auto stageA = [&](int kt, int buf) {
        const int kb = kt * BK;
        const bool fullK = (kb + BK) <= K;
        for (int c = tid; c < (BM * BK / 8); c += THREADS) {
            int row = c >> 2;
            int kc  = (c & 3) << 3;
            const bf16* src = Az + (size_t)(m0 + row) * lda + (kb + kc);
            bf16* dst = &sA[buf][row * LDSS + kc];
            if (fullK) {
#if USE_ASYNC
                __builtin_amdgcn_global_load_async_to_lds_b128(
                    (AS1 v4i*)src, (AS3 v4i*)dst, 0, 0);
#else
                v8bf val;
                __builtin_memcpy(&val, src, 16);
                *(v8bf*)dst = val;
#endif
            } else {
                v8bf val;
                for (int i = 0; i < 8; ++i)
                    val[i] = (kb + kc + i < K) ? src[i] : (bf16)0.0f;
                *(v8bf*)dst = val;
            }
        }
    };
    auto stageB = [&](int kt, int buf) {
        const int kb = kt * BK;
        const bool fullK = (kb + BK) <= K;
        if (BLAY == 1) {
            // B stored [N][K]: same orientation as sB.
            for (int c = tid; c < (BN * BK / 8); c += THREADS) {
                int row = c >> 2;
                int kc  = (c & 3) << 3;
                bf16* dst = &sB[buf][row * LDSS + kc];
                if (fullK && (n0 + row) < N) {
                    const bf16* src = Bz + (size_t)(n0 + row) * ldb + (kb + kc);
#if USE_ASYNC
                    __builtin_amdgcn_global_load_async_to_lds_b128(
                        (AS1 v4i*)src, (AS3 v4i*)dst, 0, 0);
#else
                    v8bf val;
                    __builtin_memcpy(&val, src, 16);
                    *(v8bf*)dst = val;
#endif
                } else {
                    v8bf val;
                    for (int i = 0; i < 8; ++i) {
                        int gk = kb + kc + i;
                        val[i] = ((n0 + row) < N && gk < K)
                                 ? Bz[(size_t)(n0 + row) * ldb + gk] : (bf16)0.0f;
                    }
                    *(v8bf*)dst = val;
                }
            }
        } else {
            // B stored [K][N]: transpose while staging (coalesced reads).
            int k   = tid >> 3;          // 0..31
            int nn0 = (tid & 7) << 4;    // 0,16,...,112
            bool kin = (kb + k) < K;
            if (kin && (n0 + nn0 + 16) <= N) {
                const bf16* src = Bz + (size_t)(kb + k) * ldb + (n0 + nn0);
                v8bf v0, v1;
                __builtin_memcpy(&v0, src, 16);
                __builtin_memcpy(&v1, src + 8, 16);
                for (int i = 0; i < 8; ++i) {
                    sB[buf][(nn0 + i)     * LDSS + k] = v0[i];
                    sB[buf][(nn0 + 8 + i) * LDSS + k] = v1[i];
                }
            } else {
                for (int i = 0; i < 16; ++i) {
                    int n = nn0 + i;
                    bf16 v = (bf16)0.0f;
                    if (kin && (n0 + n) < N) v = Bz[(size_t)(kb + k) * ldb + (n0 + n)];
                    sB[buf][n * LDSS + k] = v;
                }
            }
        }
    };
    // ---- compute one K-step from LDS buffer `buf` ----
    auto compute = [&](int buf) {
        // A frag (16x32 bf16): lane<16 -> M=lane, K in [0..8)+[16..24);
        //                      lane>=16 -> K in [8..16)+[24..32).
        v16bf afr[4], bfr[2];
        for (int mi = 0; mi < 4; ++mi) {
            const bf16* p = &sA[buf][(wm * 64 + mi * 16 + lm) * LDSS + lh * 8];
            union { v16bf v; v8bf h[2]; } u;
            u.h[0] = *(const v8bf*)p;
            u.h[1] = *(const v8bf*)(p + 16);
            afr[mi] = u.v;
        }
        // B frag (32x16 bf16): lane = N column, K in [0..8)+[16..24) (lane<16) or [8..16)+[24..32).
        for (int ni = 0; ni < 2; ++ni) {
            const bf16* p = &sB[buf][(wn * 32 + ni * 16 + lm) * LDSS + lh * 8];
            union { v16bf v; v8bf h[2]; } u;
            u.h[0] = *(const v8bf*)p;
            u.h[1] = *(const v8bf*)(p + 16);
            bfr[ni] = u.v;
        }
        for (int mi = 0; mi < 4; ++mi)
            for (int ni = 0; ni < 2; ++ni)
                acc[mi][ni] = __builtin_amdgcn_wmma_f32_16x16x32_bf16(
                    false, afr[mi], false, bfr[ni], (short)0, acc[mi][ni],
                    false, false);
    };

    // ---- pipelined main loop (double-buffered) ----
    stageA(0, 0);
    stageB(0, 0);
    for (int kt = 0; kt < ktiles; ++kt) {
        const int buf = kt & 1;
#if USE_ASYNC
        __builtin_amdgcn_s_wait_asynccnt(0);
#endif
        __syncthreads();   // stage(kt) complete everywhere; buf^1 free for reuse
#if USE_ASYNC
        // issue next tile's async copies, then overlap with WMMAs
        if (kt + 1 < ktiles) { stageA(kt + 1, buf ^ 1); stageB(kt + 1, buf ^ 1); }
        compute(buf);
#else
        compute(buf);
        if (kt + 1 < ktiles) { stageA(kt + 1, buf ^ 1); stageB(kt + 1, buf ^ 1); }
        if (kt + 2 < ktiles)
            __builtin_prefetch(Az + (size_t)(m0 + (tid & 127)) * lda + (size_t)(kt + 2) * BK, 0, 0);
#endif
    }

    // ---- epilogue: C/D layout: VGPR r -> M = r + (lane>=16)*8, N = lane&15 ----
    const float* biasz  = (EPI >= 1) ? (bias + (size_t)z * strideBias) : (const float*)0;
    const float* residz = RESID ? (resid + (size_t)z * strideR) : (const float*)0;
    CT* Cz = C + (size_t)z * strideC;
    for (int mi = 0; mi < 4; ++mi)
        for (int ni = 0; ni < 2; ++ni) {
            int col = n0 + wn * 32 + ni * 16 + lm;
            if (col >= N) continue;
            float bcol = (EPI >= 1) ? biasz[col] : 0.f;
            for (int r = 0; r < 8; ++r) {
                int m = m0 + wm * 64 + mi * 16 + lh * 8 + r;
                float v = acc[mi][ni][r];
                if (EPI >= 1) v += bcol;
                if (EPI == 2) v = gelu_exact(v);
                size_t crow = CPERM
                    ? (size_t)((m >> 9) * cp1 + z * cp2 + (m & 511))
                    : (size_t)m;
                if (RESID) v += residz[crow * (size_t)ldr + col];
                Cz[crow * (size_t)ldc + col] = (CT)v;
            }
        }
}

// ------------------------------- launcher ----------------------------------
extern "C" void kernel_launch(void* const* d_in, const int* in_sizes, int n_in,
                              void* d_out, int out_size, void* d_ws, size_t ws_size,
                              hipStream_t stream) {
    (void)in_sizes; (void)n_in; (void)out_size; (void)ws_size;
    const float* x        = (const float*)d_in[0];
    const float* n1w      = (const float*)d_in[1];
    const float* n1b      = (const float*)d_in[2];
    const float* n2w      = (const float*)d_in[3];
    const float* n2b      = (const float*)d_in[4];
    const float* moe_g    = (const float*)d_in[5];
    const float* slot_w   = (const float*)d_in[6];
    const float* slot_g   = (const float*)d_in[7];
    const float* exp_w1   = (const float*)d_in[8];
    const float* exp_b1   = (const float*)d_in[9];
    const float* exp_w2   = (const float*)d_in[10];
    const float* exp_b2   = (const float*)d_in[11];
    const float* mlp_w1   = (const float*)d_in[12];
    const float* mlp_b1   = (const float*)d_in[13];
    const float* mlp_w2   = (const float*)d_in[14];
    const float* mlp_b2   = (const float*)d_in[15];
    float* out = (float*)d_out;

    // ---- workspace layout ----
    char* ws = (char*)d_ws;
    size_t off = 0;
    auto A256 = [](size_t b) { return (b + 255) & ~(size_t)255; };
    size_t o_xn  = off; off += A256(4ull * 4096 * 1024 * 2);
    size_t o_xm  = off; off += A256(4ull * 512 * 1024 * 2);
    size_t o_se  = off; off += A256(32ull * 512 * 1024 * 2);
    size_t o_dt  = off; off += A256(4ull * 4096 * 4096 * 2);   // dispT, reused as MLP mid
    size_t o_cb  = off; off += A256(4ull * 4096 * 4096 * 2);   // combine
    size_t o_sl  = off; off += A256(32ull * 512 * 1024 * 2);   // slots [e][b][s][d]
    size_t o_act = off; off += A256(8ull * 2048 * 2736 * 2);   // GEGLU out, padded lda
    size_t o_ob  = off; off += A256(4ull * 4096 * 1024 * 2);   // expert outputs [b][es][d]
    size_t o_x2  = off; off += A256(4ull * 4096 * 1024 * 4);
    size_t o_h2  = off; off += A256(4ull * 4096 * 1024 * 2);
    size_t o_big = off; off += A256(4ull * 4096 * 4096 * 4);   // logits / sepre / hdn
    size_t o_ws1 = off; off += A256(1024ull * 8192 * 2);
    size_t o_we1 = off; off += A256(8ull * 1024 * 5460 * 2);
    size_t o_we2 = off; off += A256(8ull * 2730 * 1024 * 2);
    size_t o_wm1 = off; off += A256(1024ull * 4096 * 2);
    size_t o_wm2 = off; off += A256(4096ull * 1024 * 2);

    bf16*  p_xn    = (bf16*)(ws + o_xn);
    bf16*  p_xm    = (bf16*)(ws + o_xm);
    bf16*  p_se    = (bf16*)(ws + o_se);
    bf16*  p_dt    = (bf16*)(ws + o_dt);
    bf16*  p_mid   = (bf16*)(ws + o_dt);    // alias (dispT dead before MLP)
    bf16*  p_cb    = (bf16*)(ws + o_cb);
    bf16*  p_sl    = (bf16*)(ws + o_sl);
    bf16*  p_act   = (bf16*)(ws + o_act);
    bf16*  p_ob    = (bf16*)(ws + o_ob);
    float* p_x2    = (float*)(ws + o_x2);
    bf16*  p_h2    = (bf16*)(ws + o_h2);
    float* p_big   = (float*)(ws + o_big);  // logits
    float* p_sepre = (float*)(ws + o_big);  // alias (before logits)
    bf16*  p_hdn   = (bf16*)(ws + o_big);   // alias (after softmaxes)
    bf16*  p_ws1   = (bf16*)(ws + o_ws1);
    bf16*  p_we1   = (bf16*)(ws + o_we1);
    bf16*  p_we2   = (bf16*)(ws + o_we2);
    bf16*  p_wm1   = (bf16*)(ws + o_wm1);
    bf16*  p_wm2   = (bf16*)(ws + o_wm2);

    const dim3 blk(THREADS);
    const int GS = 4096;   // grid-stride block count for elementwise kernels

    // 0) weights fp32 -> bf16
    k_cvt_bf16<<<GS, blk, 0, stream>>>(slot_w, p_ws1, 1024ull * 8192);
    k_cvt_bf16<<<GS, blk, 0, stream>>>(exp_w1, p_we1, 8ull * 1024 * 5460);
    k_cvt_bf16<<<GS, blk, 0, stream>>>(exp_w2, p_we2, 8ull * 2730 * 1024);
    k_cvt_bf16<<<GS, blk, 0, stream>>>(mlp_w1, p_wm1, 1024ull * 4096);
    k_cvt_bf16<<<GS, blk, 0, stream>>>(mlp_w2, p_wm2, 4096ull * 1024);

    // 1) LN1 + RMS(moe_gamma) -> xn
    k_ln_rms<<<16384, blk, 0, stream>>>(x, n1w, n1b, moe_g, p_xn);

    // 2) group mean -> xm [b,512,1024]
    k_groupmean<<<GS, blk, 0, stream>>>(p_xn, p_xm, 4ull * 512 * 1024);

    // 3) slot embeds: sepre = xm @ slot_w   [2048 x 8192]
    k_gemm<0, 0, false, false, float><<<dim3(64, 16, 1), blk, 0, stream>>>(
        p_xm, 0, 1024, p_ws1, 0, 8192, 1,
        p_sepre, 0, 8192, nullptr, 0, nullptr, 0, 0,
        2048, 8192, 1024, 0, 0);

    // 4) RMS(slot_gamma) -> se [b][e][s][d] bf16
    k_rms_slots<<<16384, blk, 0, stream>>>(p_sepre, slot_g, p_se);

    // 5) logits[b] = xn[b] @ se[b]^T   (NT: B stored [N=es][K=d])
    k_gemm<1, 0, false, false, float><<<dim3(32, 32, 4), blk, 0, stream>>>(
        p_xn, 4096ull * 1024, 1024, p_se, 8ull * 512 * 1024, 1024, 4,
        p_big, 4096ull * 4096, 4096, nullptr, 0, nullptr, 0, 0,
        4096, 4096, 1024, 0, 0);

    // 6) dispatch = softmax over tokens -> dispT [e][b][s][n]
    k_softmax_col<<<dim3(16, 4), blk, 0, stream>>>(p_big, p_dt);
    // 7) combine = softmax over slots -> [b][n][es]
    k_softmax_row<<<16384, blk, 0, stream>>>(p_big, p_cb);

    // 8) slots[e][b][s][d] = dispT @ xn   (32 batches z = e*4 + b)
    k_gemm<0, 0, false, false, bf16><<<dim3(8, 4, 32), blk, 0, stream>>>(
        p_dt, 512ull * 4096, 4096, p_xn, 4096ull * 1024, 1024, 4,
        p_sl, 512ull * 1024, 1024, nullptr, 0, nullptr, 0, 0,
        512, 1024, 4096, 0, 0);

    // 9) expert up-proj: hdn[e] = slots[e] @ w1[e] + b1[e]   [2048 x 5460]
    k_gemm<0, 1, false, false, bf16><<<dim3(43, 16, 8), blk, 0, stream>>>(
        p_sl, 2048ull * 1024, 1024, p_we1, 1024ull * 5460, 5460, 8,
        p_hdn, 2048ull * 5460, 5460, exp_b1, 5460, nullptr, 0, 0,
        2048, 5460, 1024, 0, 0);

    // 10) GEGLU -> act [e][2048][2736-padded]
    k_geglu<<<GS, blk, 0, stream>>>(p_hdn, p_act, 8ull * 2048 * 2730);

    // 11) expert down-proj, scatter rows back to [b][es][d]:
    //     crow = (m>>9)*4096 + e*512 + (m&511)
    k_gemm<0, 1, false, true, bf16><<<dim3(8, 16, 8), blk, 0, stream>>>(
        p_act, 2048ull * 2736, 2736, p_we2, 2730ull * 1024, 1024, 8,
        p_ob, 0, 1024, exp_b2, 1024, nullptr, 0, 0,
        2048, 1024, 2730, 4096, 512);

    // 12) moe_out + residual: x2[b] = combine[b] @ out[b] + x[b]
    k_gemm<0, 0, true, false, float><<<dim3(8, 32, 4), blk, 0, stream>>>(
        p_cb, 4096ull * 4096, 4096, p_ob, 4096ull * 1024, 1024, 4,
        p_x2, 4096ull * 1024, 1024, nullptr, 0, x, 4096ull * 1024, 1024,
        4096, 1024, 4096, 0, 0);

    // 13) LN2 -> h2
    k_ln<<<16384, blk, 0, stream>>>(p_x2, n2w, n2b, p_h2);

    // 14) MLP up: mid = gelu(h2 @ mlp_w1 + b1)   [16384 x 4096]
    k_gemm<0, 2, false, false, bf16><<<dim3(32, 128, 1), blk, 0, stream>>>(
        p_h2, 0, 1024, p_wm1, 0, 4096, 1,
        p_mid, 0, 4096, mlp_b1, 0, nullptr, 0, 0,
        16384, 4096, 1024, 0, 0);

    // 15) MLP down + residual: out = mid @ mlp_w2 + b2 + x2
    k_gemm<0, 1, true, false, float><<<dim3(8, 128, 1), blk, 0, stream>>>(
        p_mid, 0, 4096, p_wm2, 0, 1024, 1,
        out, 0, 1024, mlp_b2, 0, p_x2, 0, 1024,
        16384, 1024, 4096, 0, 0);
}
